// DualGraphAttentionNetwork_58042188038485
// MI455X (gfx1250) — matrix operand order repaired and checked by reading, not compiled, not run on hardware
//
#include <hip/hip_runtime.h>
#include <math.h>

typedef _Float16 h16;
typedef __attribute__((ext_vector_type(16))) _Float16 v16h;
typedef __attribute__((ext_vector_type(8)))  _Float16 v8h;
typedef __attribute__((ext_vector_type(8)))  float    v8f;

#define B_  8
#define N_  512
#define E_  256
#define FE_ 17
#define BN_ (B_*N_)

// ---------------- wave32 reductions ----------------
__device__ __forceinline__ float wave_sum(float v){
#pragma unroll
  for(int o=16;o>0;o>>=1) v += __shfl_xor(v,o,32);
  return v;
}
__device__ __forceinline__ float wave_max(float v){
#pragma unroll
  for(int o=16;o>0;o>>=1) v = fmaxf(v, __shfl_xor(v,o,32));
  return v;
}

// ---------------- WMMA core ----------------
// 16x16 f32 tile = A(16xK f16, row-major, lda) * Bt(16xK f16, row-major, ldb)^T
// A per-lane layout (16-bit A 16x32): lane m=lane&15, half=lane>>4,
//   elems 0..15 = K kk+16*half .. +15
// B per-lane layout (K-major): col n=lane&15, elems 0..15 = K kk+16*half .. +15
__device__ __forceinline__ v8f wmma_dot(const h16* __restrict__ A, int lda,
                                        const h16* __restrict__ Bt, int ldb,
                                        int K, int lane){
  const int m    = lane & 15;
  const int half = lane >> 4;
  v8f acc = {0.f,0.f,0.f,0.f,0.f,0.f,0.f,0.f};
  const h16* arow = A  + (size_t)m * lda;
  const h16* brow = Bt + (size_t)m * ldb;   // n == lane&15
  for(int kk = 0; kk < K; kk += 32){
    v16h av = *(const v16h*)(arow + kk + 16*half);
    v16h bv = *(const v16h*)(brow + kk + 16*half);
    acc = __builtin_amdgcn_wmma_f32_16x16x32_f16(false, av, false, bv,
                                                 (short)0, acc, false, false);
  }
  return acc;
}

// ---------------- generic GEMM with fused epilogue ----------------
// C = epilogue( A[M,K] * Wt[Nc,K]^T )
// c = act(acc + bias[n]); c = gamma*c + add1 + add2; store f32 / f16
__global__ __launch_bounds__(128)
void gemm_ep_kernel(const h16* __restrict__ A, const h16* __restrict__ Wt,
                    const float* __restrict__ bias, const float* __restrict__ gammaPtr,
                    const float* __restrict__ add1, const float* __restrict__ add2,
                    int act, float* __restrict__ outF, h16* __restrict__ outH,
                    int Nc, int K){
  const int lane = threadIdx.x & 31, wv = threadIdx.x >> 5;
  const int mt = blockIdx.x, nt = blockIdx.y*4 + wv;
  v8f acc = wmma_dot(A + (size_t)mt*16*K, K, Wt + (size_t)nt*16*K, K, K, lane);
  const int half = lane>>4, nl = lane&15;
  const int ncol = nt*16 + nl;
  const float scale = gammaPtr ? *gammaPtr : 1.0f;
  const float bval  = bias ? bias[ncol] : 0.0f;
#pragma unroll
  for(int r=0;r<8;r++){
    const int mrow = mt*16 + r + 8*half;
    float c = acc[r] + bval;
    if(act==1)      c = fmaxf(c, 0.f);
    else if(act==2) c = (c > 0.f) ? c : (__expf(c) - 1.f);   // CELU(alpha=1)
    const size_t idx = (size_t)mrow*Nc + ncol;
    c = scale*c + (add1 ? add1[idx] : 0.f) + (add2 ? add2[idx] : 0.f);
    if(outF) outF[idx] = c;
    if(outH) outH[idx] = (h16)c;
  }
}

// q/k/v projection with head-split store. vt=0 -> [B,H,N,Dh]; vt=1 -> [B,H,Dh,N]
__global__ __launch_bounds__(128)
void gemm_heads_kernel(const h16* __restrict__ A, const h16* __restrict__ Wt,
                       const float* __restrict__ bias, h16* __restrict__ out,
                       int K, int H, int Dh, int vt){
  const int lane = threadIdx.x & 31, wv = threadIdx.x >> 5;
  const int mt = blockIdx.x, nt = blockIdx.y*4 + wv;
  v8f acc = wmma_dot(A + (size_t)mt*16*K, K, Wt + (size_t)nt*16*K, K, K, lane);
  const int half = lane>>4, nl = lane&15;
  const int ncol = nt*16 + nl;
  const int hh = ncol / Dh, d = ncol % Dh;
  const float bval = bias ? bias[ncol] : 0.0f;
#pragma unroll
  for(int r=0;r<8;r++){
    const int mrow = mt*16 + r + 8*half;
    const int b = mrow / N_, node = mrow % N_;
    const float c = acc[r] + bval;
    const size_t idx = vt ? ((((size_t)b*H + hh)*Dh + d)*N_ + node)
                          : ((((size_t)b*H + hh)*N_ + node)*Dh + d);
    out[idx] = (h16)c;
  }
}

// scores = q*k^T*scale (+edge bias in-place) (+leaky) (+adj mask)
__global__ __launch_bounds__(128)
void attn_scores_kernel(const h16* __restrict__ q, const h16* __restrict__ k,
                        float* __restrict__ scores, const int* __restrict__ adj,
                        int Dh, int H, float scl, int useBias, int leaky, int mask){
  const int lane = threadIdx.x & 31, wv = threadIdx.x >> 5;
  const int mt = blockIdx.x, nt = blockIdx.y*4 + wv, bh = blockIdx.z;
  const int b = bh / H;
  const h16* Ap = q + (size_t)bh*N_*Dh + (size_t)mt*16*Dh;
  const h16* Bp = k + (size_t)bh*N_*Dh + (size_t)nt*16*Dh;
  v8f acc = wmma_dot(Ap, Dh, Bp, Dh, Dh, lane);
  const int half = lane>>4, nl = lane&15;
  const int kj = nt*16 + nl;
#pragma unroll
  for(int r=0;r<8;r++){
    const int qi = mt*16 + r + 8*half;
    const size_t idx = ((size_t)bh*N_ + qi)*N_ + kj;
    float s = acc[r]*scl;
    if(useBias) s += scores[idx];
    if(leaky)   s = (s > 0.f) ? s : 0.2f*s;
    if(mask && adj[((size_t)b*N_ + qi)*N_ + kj] == 0) s = -1e9f;
    scores[idx] = s;
  }
}

// ctx = probs * v (+optional residual); probs [N,N] f16, vT [Dh,N] f16
__global__ __launch_bounds__(128)
void attn_ctx_kernel(const h16* __restrict__ probs, const h16* __restrict__ vT,
                     const float* __restrict__ addF, float* __restrict__ outF,
                     h16* __restrict__ outH, int Dh, int H){
  const int lane = threadIdx.x & 31, wv = threadIdx.x >> 5;
  const int mt = blockIdx.x, nt = blockIdx.y*4 + wv, bh = blockIdx.z;
  const int b = bh / H, hh = bh % H;
  const h16* Ap = probs + (size_t)bh*N_*N_ + (size_t)mt*16*N_;
  const h16* Bp = vT    + (size_t)bh*Dh*N_ + (size_t)nt*16*N_;
  v8f acc = wmma_dot(Ap, N_, Bp, N_, N_, lane);
  const int half = lane>>4, nl = lane&15;
  const int d = nt*16 + nl;
#pragma unroll
  for(int r=0;r<8;r++){
    const int qi = mt*16 + r + 8*half;
    const int e  = hh*Dh + d;
    const size_t idx = ((size_t)b*N_ + qi)*E_ + e;
    const float c = acc[r] + (addF ? addF[idx] : 0.f);
    if(outF) outF[idx] = c;
    if(outH) outH[idx] = (h16)c;
  }
}

// row softmax over N_ cols, f32 in -> f16 probs; one wave per row
__global__ __launch_bounds__(128)
void softmax_kernel(const float* __restrict__ s, h16* __restrict__ p, int rows){
  const int lane = threadIdx.x & 31, wv = threadIdx.x >> 5;
  const int row = blockIdx.x*4 + wv;
  if(row >= rows) return;
  const float* sr = s + (size_t)row*N_;
  float vals[16]; float mx = -3.4e38f;
#pragma unroll
  for(int i=0;i<16;i++){ vals[i] = sr[lane + 32*i]; mx = fmaxf(mx, vals[i]); }
  mx = wave_max(mx);
  float sum = 0.f;
#pragma unroll
  for(int i=0;i<16;i++){ vals[i] = __expf(vals[i] - mx); sum += vals[i]; }
  sum = wave_sum(sum);
  const float inv = 1.0f / sum;
  h16* pr = p + (size_t)row*N_;
#pragma unroll
  for(int i=0;i<16;i++) pr[lane + 32*i] = (h16)(vals[i]*inv);
}

// layernorm over E_=256, f32 in -> f16 out; one wave per row
__global__ __launch_bounds__(128)
void ln_kernel(const float* __restrict__ x, const float* __restrict__ g,
               const float* __restrict__ bta, h16* __restrict__ out, int rows){
  const int lane = threadIdx.x & 31, wv = threadIdx.x >> 5;
  const int row = blockIdx.x*4 + wv;
  if(row >= rows) return;
  const float* xr = x + (size_t)row*E_;
  float v[8]; float s = 0.f, sq = 0.f;
#pragma unroll
  for(int i=0;i<8;i++){ v[i] = xr[lane + 32*i]; s += v[i]; sq += v[i]*v[i]; }
  s = wave_sum(s); sq = wave_sum(sq);
  const float mean = s * (1.0f/E_);
  const float var  = sq * (1.0f/E_) - mean*mean;
  const float rstd = rsqrtf(var + 1e-5f);
  h16* o = out + (size_t)row*E_;
#pragma unroll
  for(int i=0;i<8;i++){
    const int j = lane + 32*i;
    o[j] = (h16)((v[i]-mean)*rstd*g[j] + bta[j]);
  }
}

// f32 W[K,Nc] -> f16 transposed Wt[Nc,K]
__global__ void convw_kernel(const float* __restrict__ W, h16* __restrict__ Wt,
                             int K, int Nc){
  const int t = blockIdx.x*256 + threadIdx.x;
  if(t >= K*Nc) return;
  const int n = t / K, kk = t % K;
  Wt[t] = (h16)W[(size_t)kk*Nc + n];
}

// fused per-edge MLP: LN(17) -> relu(17x34) -> 34xH, write bias into scores[B,H,N,N]
__global__ __launch_bounds__(256)
void edge_bias_kernel(const float* __restrict__ e, const float* __restrict__ eg,
                      const float* __restrict__ eb, const float* __restrict__ w1,
                      const float* __restrict__ b1, const float* __restrict__ w2,
                      const float* __restrict__ b2, float* __restrict__ bias, int H){
  __shared__ float s_w1[FE_*2*FE_];
  __shared__ float s_b1[2*FE_];
  __shared__ float s_w2[2*FE_*4];
  __shared__ float s_b2[4];
  __shared__ float s_g[FE_], s_bb[FE_];
  for(int t=threadIdx.x; t<FE_*2*FE_; t+=256) s_w1[t]=w1[t];
  for(int t=threadIdx.x; t<2*FE_;     t+=256) s_b1[t]=b1[t];
  for(int t=threadIdx.x; t<2*FE_*H;   t+=256) s_w2[t]=w2[t];
  for(int t=threadIdx.x; t<H;         t+=256) s_b2[t]=b2[t];
  for(int t=threadIdx.x; t<FE_;       t+=256){ s_g[t]=eg[t]; s_bb[t]=eb[t]; }
  __syncthreads();
  const size_t t = (size_t)blockIdx.x*256 + threadIdx.x;     // over B*N*N
  const int b = (int)(t/((size_t)N_*N_));
  const int rem = (int)(t%((size_t)N_*N_));
  const int i = rem / N_, j = rem % N_;
  const float* er = e + t*FE_;
  float en[FE_]; float s = 0.f;
#pragma unroll
  for(int f=0; f<FE_; f++){ en[f]=er[f]; s+=en[f]; }
  const float mean = s*(1.0f/FE_); float var = 0.f;
#pragma unroll
  for(int f=0; f<FE_; f++){ const float d=en[f]-mean; var+=d*d; }
  const float rstd = rsqrtf(var*(1.0f/FE_) + 1e-5f);
#pragma unroll
  for(int f=0; f<FE_; f++) en[f] = (en[f]-mean)*rstd*s_g[f] + s_bb[f];
  float h1[2*FE_];
#pragma unroll
  for(int c=0;c<2*FE_;c++){
    float a = s_b1[c];
#pragma unroll
    for(int f=0; f<FE_; f++) a += en[f]*s_w1[f*2*FE_+c];
    h1[c] = fmaxf(a, 0.f);
  }
  for(int hh=0; hh<H; hh++){
    float o = s_b2[hh];
#pragma unroll
    for(int c=0;c<2*FE_;c++) o += h1[c]*s_w2[c*H+hh];
    bias[(((size_t)b*H+hh)*N_ + i)*N_ + j] = o;
  }
}

// ---------------- host orchestration ----------------
extern "C" void kernel_launch(void* const* d_in, const int* in_sizes, int n_in,
                              void* d_out, int out_size, void* d_ws, size_t ws_size,
                              hipStream_t stream) {
  (void)in_sizes; (void)n_in; (void)out_size; (void)ws_size;
  const float* node = (const float*)d_in[0];
  const float* edge = (const float*)d_in[1];
  const int*   adj  = (const int*)d_in[2];

  char* base = (char*)d_ws; size_t off = 0;
  auto allocB = [&](size_t bytes)->void*{ void* p = base + off; off += (bytes + 255) & ~(size_t)255; return p; };
  auto aH = [&](size_t n)->h16*  { return (h16*)  allocB(n*sizeof(h16));  };
  auto aF = [&](size_t n)->float*{ return (float*)allocB(n*sizeof(float)); };

  // transposed f16 weights (reconverted each layer)
  h16 *w_proj=aH((size_t)E_*E_), *w_wq=aH((size_t)E_*E_), *w_wk=aH((size_t)E_*E_), *w_wv=aH((size_t)E_*E_);
  h16 *w_ffn1=aH((size_t)E_*2*E_), *w_ffn2=aH((size_t)2*E_*E_);
  h16 *w_mq=aH((size_t)E_*E_), *w_mk=aH((size_t)E_*E_), *w_mv=aH((size_t)E_*E_), *w_mo=aH((size_t)E_*E_);
  h16 *w_m1=aH((size_t)E_*E_), *w_m2=aH((size_t)E_*E_);
  // activations
  const size_t NE = (size_t)BN_*E_;
  h16  *xbf=aH(NE), *xp_h=aH(NE), *qb=aH(NE), *kb=aH(NE), *vtb=aH(NE);
  h16  *hn=aH(NE), *fmid=aH((size_t)BN_*2*E_), *xl_h=aH(NE), *oin=aH(NE), *gn=aH(NE), *mln=aH(NE);
  float*xp_f=aF(NE), *hbuf=aF(NE), *xl_f=aF(NE), *fused=aF(NE), *mmid=aF(NE), *xcur=aF(NE);
  float*scores=aF((size_t)B_*4*N_*N_);
  h16  *probs=aH((size_t)B_*4*N_*N_);

  hipMemcpyAsync(xcur, node, NE*sizeof(float), hipMemcpyDeviceToDevice, stream);

  const dim3 blk(128);
  const dim3 gEE(BN_/16, E_/64);        // [BN x E] GEMMs
  const dim3 gE2(BN_/16, 2*E_/64);      // [BN x 2E]

  for(int L=0; L<3; L++){
    const float* P[36];
    for(int i=0;i<36;i++) P[i] = (const float*)d_in[3 + L*36 + i];
    const int H = (L<2)?4:1, Dh = E_/H, leaky = (L<2)?1:0;
    const float scl = 1.0f / sqrtf((float)Dh);
    const int cEE = (E_*E_+255)/256, cE2 = (E_*2*E_+255)/256;

    convw_kernel<<<cEE,256,0,stream>>>(P[2],  w_proj, E_, E_);
    convw_kernel<<<cEE,256,0,stream>>>(P[4],  w_wq,   E_, E_);
    convw_kernel<<<cEE,256,0,stream>>>(P[5],  w_wk,   E_, E_);
    convw_kernel<<<cEE,256,0,stream>>>(P[6],  w_wv,   E_, E_);
    convw_kernel<<<cE2,256,0,stream>>>(P[13], w_ffn1, E_, 2*E_);
    convw_kernel<<<cE2,256,0,stream>>>(P[15], w_ffn2, 2*E_, E_);
    convw_kernel<<<cEE,256,0,stream>>>(P[19], w_mq,   E_, E_);
    convw_kernel<<<cEE,256,0,stream>>>(P[21], w_mk,   E_, E_);
    convw_kernel<<<cEE,256,0,stream>>>(P[23], w_mv,   E_, E_);
    convw_kernel<<<cEE,256,0,stream>>>(P[25], w_mo,   E_, E_);
    convw_kernel<<<cEE,256,0,stream>>>(P[30], w_m1,   E_, E_);
    convw_kernel<<<cEE,256,0,stream>>>(P[34], w_m2,   E_, E_);

    // ---- GAT ----
    ln_kernel<<<BN_/4,blk,0,stream>>>(xcur, P[0], P[1], xbf, BN_);
    gemm_ep_kernel<<<gEE,blk,0,stream>>>(xbf, w_proj, P[3], nullptr,nullptr,nullptr, 0, xp_f, xp_h, E_, E_);
    gemm_heads_kernel<<<gEE,blk,0,stream>>>(xp_h, w_wq, nullptr, qb,  E_, H, Dh, 0);
    gemm_heads_kernel<<<gEE,blk,0,stream>>>(xp_h, w_wk, nullptr, kb,  E_, H, Dh, 0);
    gemm_heads_kernel<<<gEE,blk,0,stream>>>(xp_h, w_wv, nullptr, vtb, E_, H, Dh, 1);
    edge_bias_kernel<<<(B_*N_*N_)/256,256,0,stream>>>(edge, P[7],P[8],P[9],P[10],P[11],P[12], scores, H);
    dim3 gs(N_/16, N_/64, B_*H);
    attn_scores_kernel<<<gs,blk,0,stream>>>(qb, kb, scores, adj, Dh, H, scl, 1, leaky, 1);
    softmax_kernel<<<(B_*H*N_)/4,blk,0,stream>>>(scores, probs, B_*H*N_);
    dim3 gc(N_/16, Dh/64, B_*H);
    attn_ctx_kernel<<<gc,blk,0,stream>>>(probs, vtb, xp_f, hbuf, nullptr, Dh, H);   // h = xp + ctx
    ln_kernel<<<BN_/4,blk,0,stream>>>(hbuf, P[17], P[18], hn, BN_);
    gemm_ep_kernel<<<gE2,blk,0,stream>>>(hn, w_ffn1, P[14], nullptr,nullptr,nullptr, 1, nullptr, fmid, 2*E_, E_);
    gemm_ep_kernel<<<gEE,blk,0,stream>>>(fmid, w_ffn2, P[16], nullptr, hbuf, nullptr, 0, xl_f, xl_h, E_, 2*E_); // xl = h + f

    // ---- MHA ----
    gemm_heads_kernel<<<gEE,blk,0,stream>>>(xl_h, w_mq, P[20], qb,  E_, H, Dh, 0);
    gemm_heads_kernel<<<gEE,blk,0,stream>>>(xl_h, w_mk, P[22], kb,  E_, H, Dh, 0);
    gemm_heads_kernel<<<gEE,blk,0,stream>>>(xl_h, w_mv, P[24], vtb, E_, H, Dh, 1);
    attn_scores_kernel<<<gs,blk,0,stream>>>(qb, kb, scores, adj, Dh, H, scl, 0, 0, 0);
    softmax_kernel<<<(B_*H*N_)/4,blk,0,stream>>>(scores, probs, B_*H*N_);
    attn_ctx_kernel<<<gc,blk,0,stream>>>(probs, vtb, nullptr, nullptr, oin, Dh, H);
    // fused = x + xl + gamma*(oin@oW + ob)
    gemm_ep_kernel<<<gEE,blk,0,stream>>>(oin, w_mo, P[26], P[27], xcur, xl_f, 0, fused, nullptr, E_, E_);

    // ---- MLP head ----
    ln_kernel<<<BN_/4,blk,0,stream>>>(fused, P[28], P[29], gn, BN_);
    gemm_ep_kernel<<<gEE,blk,0,stream>>>(gn, w_m1, P[31], nullptr,nullptr,nullptr, 2, mmid, nullptr, E_, E_); // celu
    ln_kernel<<<BN_/4,blk,0,stream>>>(mmid, P[32], P[33], mln, BN_);
    float* outp = (L==2) ? (float*)d_out : xcur;
    gemm_ep_kernel<<<gEE,blk,0,stream>>>(mln, w_m2, P[35], nullptr,nullptr,nullptr, 0, outp, nullptr, E_, E_);
  }
}
